// simpleAttention_39968965657155
// MI455X (gfx1250) — matrix operand order repaired and checked by reading, not compiled, not run on hardware
//
#include <hip/hip_runtime.h>
#include <hip/hip_bf16.h>
#include <cstdint>
#include <cstddef>

typedef __attribute__((ext_vector_type(16))) __bf16 v16bf;
typedef __attribute__((ext_vector_type(8)))  float  v8f;

#define B_ 16
#define E_ 128
#define M_ 64
#define D_ 256

// LDS strides (elements)
#define KV_S  264   // bf16, 64 rows   (528B rows -> bank-spread for A-frag loads)
#define KP_S  264   // bf16, 64 rows
#define VPT_S 72    // bf16, 256 rows  (144B rows, 16B aligned chunks)
#define PT_S  72    // bf16, 128 rows
#define DT_S  66    // f32, 128 rows   (264B rows -> conflict-free column softmax)

union ABFrag { v16bf v; uint4 q[2]; };
union PK8    { __bf16 h[8]; uint4 u; };
union PK4    { __bf16 h[4]; uint2 u; };

__global__ __launch_bounds__(256) void prep_weights(const float* __restrict__ Wk,
                                                    const float* __restrict__ Wv,
                                                    __bf16* __restrict__ wkb,
                                                    __bf16* __restrict__ wvb) {
    int i = blockIdx.x * 256 + threadIdx.x;   // 65536 threads total
    wkb[i] = (__bf16)Wk[i];
    wvb[i] = (__bf16)Wv[i];
}

// qp[row][k] = sum_d q[row][d] * Wq[k][d]   (row = b*E + f), stored bf16 natural layout
__global__ __launch_bounds__(256) void prep_qp(const float* __restrict__ q,
                                               const float* __restrict__ Wq,
                                               __bf16* __restrict__ qpb) {
    __shared__ float qrow[D_];
    const int row = blockIdx.x;      // 0..2047
    const int k   = threadIdx.x;     // 0..255
    qrow[k] = q[(size_t)row * D_ + k];
    __syncthreads();
    const float* w = Wq + (size_t)k * D_;
    float acc = 0.f;
    for (int d = 0; d < D_; d += 4) {
        float4 wv4 = *(const float4*)(w + d);
        acc = fmaf(qrow[d + 0], wv4.x, acc);
        acc = fmaf(qrow[d + 1], wv4.y, acc);
        acc = fmaf(qrow[d + 2], wv4.z, acc);
        acc = fmaf(qrow[d + 3], wv4.w, acc);
    }
    qpb[(size_t)row * D_ + k] = (__bf16)acc;
}

__global__ __launch_bounds__(256) void attn_main(const float* __restrict__ kv,
                                                 const float* __restrict__ mask,
                                                 const __bf16* __restrict__ qpb,
                                                 const __bf16* __restrict__ wkb,
                                                 const __bf16* __restrict__ wvb,
                                                 float* __restrict__ out_value,
                                                 float* __restrict__ out_dots) {
    // 33792 (s_kv | s_dotsT) + 33792 (s_kp) + 36864 (s_vpT) + 18432 (s_pT) + 256 (mask)
    __shared__ __align__(16) unsigned char smem[123136];
    __bf16* s_kv    = (__bf16*)(smem);            // [64][264] bf16  (phase 1-2)
    float*  s_dotsT = (float*)(smem);             // [128][66] f32   (phase 3-4, aliases s_kv)
    __bf16* s_kp    = (__bf16*)(smem + 33792);    // [64][264] bf16
    __bf16* s_vpT   = (__bf16*)(smem + 67584);    // [256][72] bf16  (vp transposed)
    __bf16* s_pT    = (__bf16*)(smem + 104448);   // [128][72] bf16  (probs transposed)
    float*  s_mask  = (float*)(smem + 122880);    // [64] f32

    const int tid  = threadIdx.x;
    const int wv   = tid >> 5;      // wave id 0..7
    const int lane = tid & 31;
    const int lo   = lane & 15;
    const int hi   = lane >> 4;
    const int be   = blockIdx.x;    // b*E + e
    const int b    = be >> 7;

    // ---- Phase 1: kv[b,e] (64x256 f32) -> bf16 LDS; load mention mask ----
    {
        const float* kvp = kv + (size_t)be * (M_ * D_);
        for (int i = tid; i < (M_ * D_) / 4; i += 256) {
            int idx = i * 4;
            int m = idx >> 8;          // /256
            int k = idx & 255;
            float4 v = *(const float4*)(kvp + idx);
            PK4 p;
            p.h[0] = (__bf16)v.x; p.h[1] = (__bf16)v.y;
            p.h[2] = (__bf16)v.z; p.h[3] = (__bf16)v.w;
            *(uint2*)(s_kv + m * KV_S + k) = p.u;
        }
        if (tid < M_) s_mask[tid] = mask[(size_t)be * M_ + tid];
    }
    __syncthreads();

    // ---- Phase 2: kp = kv @ Wk^T, vp = kv @ Wv^T (fused, shared A fragments) ----
    // Output tiles: 4 (m) x 16 (n) ; K-steps: 8 (d=256)
    for (int t = wv; t < 64; t += 8) {
        const int mt = t >> 4, nt = t & 15;
        v8f acck = {};
        v8f accv = {};
        const __bf16* arow  = s_kv + (mt * 16 + lo) * KV_S;
        const __bf16* bkrow = wkb + (size_t)(nt * 16 + lo) * D_;   // Wk natural = B^T
        const __bf16* bvrow = wvb + (size_t)(nt * 16 + lo) * D_;
        for (int ks = 0; ks < 8; ++ks) {
            ABFrag a, bk, bb;
            const __bf16* ap = arow + ks * 32 + hi * 8;
            a.q[0] = *(const uint4*)(ap);
            a.q[1] = *(const uint4*)(ap + 16);
            const __bf16* bkp = bkrow + ks * 32 + hi * 8;
            bk.q[0] = *(const uint4*)(bkp);
            bk.q[1] = *(const uint4*)(bkp + 16);
            const __bf16* bvp = bvrow + ks * 32 + hi * 8;
            bb.q[0] = *(const uint4*)(bvp);
            bb.q[1] = *(const uint4*)(bvp + 16);
            acck = __builtin_amdgcn_wmma_f32_16x16x32_bf16(false, a.v, false, bk.v,
                                                           (short)0, acck, false, false);
            accv = __builtin_amdgcn_wmma_f32_16x16x32_bf16(false, a.v, false, bb.v,
                                                           (short)0, accv, false, false);
        }
        // kp tile -> row-major LDS (A operand for GEMM2): col fixed per lane, rows scattered
        __bf16* kpc = s_kp + (mt * 16 + 8 * hi) * KP_S + (nt * 16 + lo);
        #pragma unroll
        for (int r = 0; r < 8; ++r) kpc[r * KP_S] = (__bf16)acck[r];
        // vp tile -> transposed LDS [v][m] (B^T operand for GEMM3): contiguous per lane
        PK8 pv;
        #pragma unroll
        for (int r = 0; r < 8; ++r) pv.h[r] = (__bf16)accv[r];
        *(uint4*)(s_vpT + (nt * 16 + lo) * VPT_S + mt * 16 + 8 * hi) = pv.u;
    }
    __syncthreads();

    // ---- Phase 3: dots[m,f] = sum_k kp[m,k]*qp[b,f,k] ; store transposed to s_dotsT[f][m] ----
    {
        const __bf16* qpbb = qpb + (size_t)b * (E_ * D_);   // qp natural [f][k] = B^T
        for (int t = wv; t < 32; t += 8) {
            const int mt = t >> 3, ft = t & 7;
            v8f acc = {};
            const __bf16* arow = s_kp + (mt * 16 + lo) * KP_S;
            const __bf16* brow = qpbb + (size_t)(ft * 16 + lo) * D_;
            for (int ks = 0; ks < 8; ++ks) {
                ABFrag a, bb;
                const __bf16* ap = arow + ks * 32 + hi * 8;
                a.q[0] = *(const uint4*)(ap);
                a.q[1] = *(const uint4*)(ap + 16);
                const __bf16* bp = brow + ks * 32 + hi * 8;
                bb.q[0] = *(const uint4*)(bp);
                bb.q[1] = *(const uint4*)(bp + 16);
                acc = __builtin_amdgcn_wmma_f32_16x16x32_bf16(false, a.v, false, bb.v,
                                                              (short)0, acc, false, false);
            }
            // D frag: rows m = mt*16+8*hi+r, col f = ft*16+lo -> contiguous along m in s_dotsT
            float* dp = s_dotsT + (ft * 16 + lo) * DT_S + mt * 16 + 8 * hi;
            #pragma unroll
            for (int r = 0; r < 8; r += 2) {
                float2 pr; pr.x = acc[r]; pr.y = acc[r + 1];
                *(float2*)(dp + r) = pr;
            }
        }
    }
    __syncthreads();

    // ---- Phase 4: softmax over m per column f, threshold, emit dots + bf16 probs ----
    if (tid < E_) {
        const int f = tid;
        float* row = s_dotsT + f * DT_S;
        float mx = -3.0e38f;
        for (int m = 0; m < M_; ++m) {
            float v = row[m] * 0.0625f + s_mask[m];   // scale = 256^-0.5
            row[m] = v;
            mx = fmaxf(mx, v);
        }
        float sum = 0.f;
        for (int m = 0; m < M_; ++m) {
            float e = __expf(row[m] - mx);
            row[m] = e;
            sum += e;
        }
        const float inv = 1.f / sum;
        float* od = out_dots + ((size_t)be * E_ + f) * M_;
        __bf16* pr = s_pT + f * PT_S;
        for (int m = 0; m < M_; ++m) {
            float p = row[m] * inv;
            p = (p >= 0.01f) ? p : 0.f;
            od[m] = p;
            pr[m] = (__bf16)p;
        }
    }
    __syncthreads();

    // ---- Phase 5: value[f,v] = sum_m pT[f,m] * vp[m,v]  (K=64 -> 2 K-steps) ----
    for (int t = wv; t < 128; t += 8) {
        const int ft = t >> 4, vt = t & 15;
        v8f acc = {};
        const __bf16* arow = s_pT + (ft * 16 + lo) * PT_S;
        const __bf16* brow = s_vpT + (vt * 16 + lo) * VPT_S;
        #pragma unroll
        for (int ks = 0; ks < 2; ++ks) {
            ABFrag a, bb;
            const __bf16* ap = arow + ks * 32 + hi * 8;
            a.q[0] = *(const uint4*)(ap);
            a.q[1] = *(const uint4*)(ap + 16);
            const __bf16* bp = brow + ks * 32 + hi * 8;
            bb.q[0] = *(const uint4*)(bp);
            bb.q[1] = *(const uint4*)(bp + 16);
            acc = __builtin_amdgcn_wmma_f32_16x16x32_bf16(false, a.v, false, bb.v,
                                                          (short)0, acc, false, false);
        }
        float* vo = out_value + ((size_t)be * E_ + ft * 16 + 8 * hi) * D_ + vt * 16 + lo;
        #pragma unroll
        for (int r = 0; r < 8; ++r) vo[(size_t)r * D_] = acc[r];
    }
}

extern "C" void kernel_launch(void* const* d_in, const int* in_sizes, int n_in,
                              void* d_out, int out_size, void* d_ws, size_t ws_size,
                              hipStream_t stream) {
    const float* q    = (const float*)d_in[0];  // [B,E,D]
    const float* kv   = (const float*)d_in[1];  // [B,E,M,D]
    const float* mask = (const float*)d_in[2];  // [B,E,M,1]
    const float* Wq   = (const float*)d_in[3];  // [D,D]
    const float* Wk   = (const float*)d_in[4];  // [D,D]
    const float* Wv   = (const float*)d_in[5];  // [D,D]

    __bf16* qpb = (__bf16*)d_ws;                                    // 2048*256*2 = 1 MB
    __bf16* wkb = (__bf16*)((char*)d_ws + 1048576);                 // 128 KB
    __bf16* wvb = (__bf16*)((char*)d_ws + 1048576 + 131072);        // 128 KB

    float* out_value = (float*)d_out;                               // [B,E,E,D]
    float* out_dots  = out_value + (size_t)B_ * E_ * E_ * D_;       // [B,E,E,M]

    prep_weights<<<(D_ * D_) / 256, 256, 0, stream>>>(Wk, Wv, wkb, wvb);
    prep_qp<<<B_ * E_, 256, 0, stream>>>(q, Wq, qpb);
    attn_main<<<B_ * E_, 256, 0, stream>>>(kv, mask, qpb, wkb, wvb, out_value, out_dots);
}
